// GPT_43327630082615
// MI455X (gfx1250) — compile-verified, matrix operand order checked by reading, not run on hardware
//
#include <hip/hip_runtime.h>
#include <hip/hip_bf16.h>
#include <math.h>
#include <stdint.h>

// ---------------- model constants ----------------
constexpr int TT   = 2048;
constexpr int DD   = 768;
constexpr int HH   = 6;
constexpr int HDIM = 128;
constexpr int VV   = 50257;
constexpr int VPAD = 50304;
constexpr float EPSF = 1.1920929e-07f;
constexpr float ATTN_SCALE = 0.12f;
// scores bounded: |s| <= 0.12*||q||*||k|| = 0.12*128 = 15.36
// => exp(s - SMAX_SHIFT) <= exp(9.36) ~ 1.16e4 < 65504 (f16 max): no overflow.
constexpr float SMAX_SHIFT = 6.0f;

typedef __attribute__((ext_vector_type(16))) _Float16 v16h;
typedef __attribute__((ext_vector_type(8)))  _Float16 v8h;
typedef __attribute__((ext_vector_type(8)))  float    v8f;
typedef __attribute__((ext_vector_type(4)))  float    v4f;

// Build a 16-half WMMA fragment from two contiguous 16-byte runs.
__device__ __forceinline__ v16h frag16(const _Float16* p0, const _Float16* p1) {
    v8h a = *(const v8h*)p0;
    v8h b = *(const v8h*)p1;
    return __builtin_shufflevector(a, b, 0,1,2,3,4,5,6,7,8,9,10,11,12,13,14,15);
}

// =====================================================================
// 1) Block-mask build: flag 0=skip, 1=full, 2=partial (causal & doc)
// =====================================================================
__global__ void mask_kernel(const int* __restrict__ seq, const int* __restrict__ swp,
                            int* __restrict__ docs, unsigned char* __restrict__ bflags) {
    if (threadIdx.x != 0 || blockIdx.x != 0) return;
    int c = 0;
    for (int t = 0; t < TT; ++t) { c += (seq[t] == 50256) ? 1 : 0; docs[t] = c; }
    int dlo[16], dhi[16];
    for (int b = 0; b < 16; ++b) { dlo[b] = docs[b * 128]; dhi[b] = docs[b * 128 + 127]; }
    for (int variant = 0; variant < 2; ++variant) {
        int sw = (variant == 0) ? swp[0] : (swp[0] / 2);
        for (int q = 0; q < 16; ++q) {
            int allm[16], partm[16], fullcnt = 0;
            for (int k = 0; k < 16; ++k) {
                int anyb = (q >= k) && (dlo[q] <= dhi[k]) && (dhi[q] >= dlo[k]);
                int allb = (q > k) && (dlo[q] == dhi[k]) && (dhi[q] == dlo[k]);
                int part = anyb && !allb;
                allm[k] = allb; partm[k] = part; fullcnt += allb;
            }
            int prank[16], frank[16], cp = 0, cf = 0;
            for (int k = 15; k >= 0; --k) { prank[k] = cp; frank[k] = cf; cp += partm[k]; cf += allm[k]; }
            int limp = sw - fullcnt; if (limp < 1) limp = 1;
            for (int k = 0; k < 16; ++k) {
                unsigned char flag = 0;
                if (allm[k] && frank[k] < sw - 1) flag = 1;
                else if (partm[k] && prank[k] < limp) flag = 2;
                bflags[variant * 256 + q * 16 + k] = flag;
            }
        }
    }
}

// =====================================================================
// 2) Embedding gather + rms_norm -> x, x0 (fp32)
// =====================================================================
__global__ void embed_kernel(const int* __restrict__ seq, const float* __restrict__ embed_w,
                             float* __restrict__ x, float* __restrict__ x0) {
    __shared__ float red[256];
    const int t = blockIdx.x, j = threadIdx.x;
    const size_t base = (size_t)seq[t] * DD;
    float e0 = embed_w[base + j], e1 = embed_w[base + j + 256], e2 = embed_w[base + j + 512];
    red[j] = e0 * e0 + e1 * e1 + e2 * e2;
    __syncthreads();
    for (int s = 128; s > 0; s >>= 1) { if (j < s) red[j] += red[j + s]; __syncthreads(); }
    float scale = rsqrtf(red[0] * (1.0f / DD) + EPSF);
    float o0 = e0 * scale, o1 = e1 * scale, o2 = e2 * scale;
    size_t r = (size_t)t * DD;
    x[r + j] = o0; x[r + j + 256] = o1; x[r + j + 512] = o2;
    x0[r + j] = o0; x0[r + j + 256] = o1; x0[r + j + 512] = o2;
}

__global__ void rmsnorm_f16_kernel(const float* __restrict__ x, _Float16* __restrict__ out) {
    __shared__ float red[256];
    const int t = blockIdx.x, j = threadIdx.x;
    const size_t r = (size_t)t * DD;
    float e0 = x[r + j], e1 = x[r + j + 256], e2 = x[r + j + 512];
    red[j] = e0 * e0 + e1 * e1 + e2 * e2;
    __syncthreads();
    for (int s = 128; s > 0; s >>= 1) { if (j < s) red[j] += red[j + s]; __syncthreads(); }
    float scale = rsqrtf(red[0] * (1.0f / DD) + EPSF);
    out[r + j]       = (_Float16)(e0 * scale);
    out[r + j + 256] = (_Float16)(e1 * scale);
    out[r + j + 512] = (_Float16)(e2 * scale);
}

// =====================================================================
// 3) Tiled WMMA GEMM:  C[M,N] = A[M,K](f16) * B[N,K]^T(f32 weights)
//    Double-buffered LDS ping-pong: next A tile via async global->LDS,
//    next B tile via registers (fp32->f16 cvt); ONE barrier per K-step.
//    FUSE: 0 store f32 | 1 f16 relu^2 | 2 += f32 | 3 lm-head loss epilogue
// =====================================================================
template <int FUSE>
__global__ __launch_bounds__(128)
void gemm_wmma(const _Float16* __restrict__ A, const float* __restrict__ B,
               float* __restrict__ C, _Float16* __restrict__ Ch,
               float* __restrict__ rowsum, float* __restrict__ ztarget,
               const int* __restrict__ target,
               int M, int N, int K, float lm_inv) {
    __shared__ _Float16 As[2][64 * 32];
    __shared__ _Float16 Bs[2][64 * 32];
    const int tid  = threadIdx.x;
    const int wave = tid >> 5;
    const int lane = tid & 31;
    const int bm = blockIdx.y * 64;
    const int bn = blockIdx.x * 64;

    v8f acc[4];
#pragma unroll
    for (int nt = 0; nt < 4; ++nt)
#pragma unroll
        for (int r = 0; r < 8; ++r) acc[nt][r] = 0.0f;

    const int khalf = lane >> 4;
    const int arow  = wave * 16 + (lane & 15);
    const int srow  = tid >> 1;
    const int scol  = (tid & 1) * 16;
    const int nk    = K >> 5;

    v4f bf[4];  // in-flight B tile (16 floats / thread)

    auto issueA = [&](int kt, int buf) {  // async DMA straight into LDS
        const _Float16* src = A + (size_t)(bm + srow) * K + kt * 32 + scol;
        unsigned l0 = (unsigned)(size_t)&As[buf][srow * 32 + scol];
        asm volatile("global_load_async_to_lds_b128 %0, %1, off"
                     :: "v"(l0), "v"(src) : "memory");
        unsigned l1 = (unsigned)(size_t)&As[buf][srow * 32 + scol + 8];
        asm volatile("global_load_async_to_lds_b128 %0, %1, off"
                     :: "v"(l1), "v"(src + 8) : "memory");
    };
    auto loadB = [&](int kt) {
        const float* src = B + (size_t)(bn + srow) * K + kt * 32 + scol;
        bf[0] = *(const v4f*)src;        bf[1] = *(const v4f*)(src + 4);
        bf[2] = *(const v4f*)(src + 8);  bf[3] = *(const v4f*)(src + 12);
    };
    auto storeB = [&](int buf) {
        v8h h0, h1;
#pragma unroll
        for (int i = 0; i < 4; ++i) {
            h0[i] = (_Float16)bf[0][i]; h0[4 + i] = (_Float16)bf[1][i];
            h1[i] = (_Float16)bf[2][i]; h1[4 + i] = (_Float16)bf[3][i];
        }
        *(v8h*)&Bs[buf][srow * 32 + scol]     = h0;
        *(v8h*)&Bs[buf][srow * 32 + scol + 8] = h1;
    };

    // prologue: tile 0 into buffer 0
    issueA(0, 0);
    loadB(0);
    storeB(0);
    asm volatile("s_wait_asynccnt 0" ::: "memory");
    __syncthreads();

    for (int kt = 0; kt < nk; ++kt) {
        const int cur = kt & 1, nxt = cur ^ 1;
        if (kt + 1 < nk) {
            issueA(kt + 1, nxt);     // lands in back buffer during compute
            loadB(kt + 1);           // in-flight in VGPRs during compute
        }
        const v16h a = frag16(&As[cur][arow * 32 + khalf * 8],
                              &As[cur][arow * 32 + 16 + khalf * 8]);
#pragma unroll
        for (int nt = 0; nt < 4; ++nt) {
            const int bcol = nt * 16 + (lane & 15);
            const v16h b = frag16(&Bs[cur][bcol * 32 + khalf * 8],
                                  &Bs[cur][bcol * 32 + 16 + khalf * 8]);
            acc[nt] = __builtin_amdgcn_wmma_f32_16x16x32_f16(
                false, a, false, b, (short)0, acc[nt], false, false);
        }
        if (kt + 1 < nk) storeB(nxt);
        asm volatile("s_wait_asynccnt 0" ::: "memory");
        __syncthreads();             // single barrier per K-step
    }

    const int rowbase = bm + wave * 16 + khalf * 8;
    if (FUSE == 3) {
#pragma unroll
        for (int r = 0; r < 8; ++r) {
            const int row = rowbase + r;
            float es = 0.0f;
#pragma unroll
            for (int nt = 0; nt < 4; ++nt) {
                const int col = bn + nt * 16 + (lane & 15);
                const float z = 30.0f / (1.0f + __expf(-acc[nt][r] * lm_inv));
                es += __expf(z);
                if (col == target[row]) ztarget[row] = z;
            }
            es += __shfl_xor(es, 1, 32);
            es += __shfl_xor(es, 2, 32);
            es += __shfl_xor(es, 4, 32);
            es += __shfl_xor(es, 8, 32);
            if ((lane & 15) == 0) atomicAdd(&rowsum[row], es);
        }
    } else {
#pragma unroll
        for (int nt = 0; nt < 4; ++nt) {
            const int col = bn + nt * 16 + (lane & 15);
#pragma unroll
            for (int r = 0; r < 8; ++r) {
                const int row = rowbase + r;
                const float v = acc[nt][r];
                if (FUSE == 0) {
                    C[(size_t)row * N + col] = v;
                } else if (FUSE == 1) {
                    float rv = v > 0.0f ? v : 0.0f;
                    Ch[(size_t)row * N + col] = (_Float16)(rv * rv);
                } else {
                    C[(size_t)row * N + col] += v;
                }
            }
        }
    }
}

// =====================================================================
// 4) QK rms_norm + rotary + V lambda/ve fusion -> q/k/v (H,T,HD) f16
// =====================================================================
__global__ void rope_kernel(const float* __restrict__ qkv, const float* __restrict__ attn_lam,
                            int layer, const float* __restrict__ ve_w, int vei,
                            const int* __restrict__ seq,
                            _Float16* __restrict__ q16, _Float16* __restrict__ k16,
                            _Float16* __restrict__ v16) {
    __shared__ float red[128];
    __shared__ float qsh[128], ksh[128];
    __shared__ float rq_s, rk_s;
    const int t = blockIdx.x, h = blockIdx.y, d = threadIdx.x;
    const size_t rb = (size_t)t * (3 * DD);
    const float qv = qkv[rb + h * HDIM + d];
    const float kv = qkv[rb + DD + h * HDIM + d];
    const float vv = qkv[rb + 2 * DD + h * HDIM + d];

    red[d] = qv * qv; __syncthreads();
    for (int s = 64; s > 0; s >>= 1) { if (d < s) red[d] += red[d + s]; __syncthreads(); }
    if (d == 0) rq_s = rsqrtf(red[0] * (1.0f / HDIM) + EPSF);
    __syncthreads();
    red[d] = kv * kv; __syncthreads();
    for (int s = 64; s > 0; s >>= 1) { if (d < s) red[d] += red[d + s]; __syncthreads(); }
    if (d == 0) rk_s = rsqrtf(red[0] * (1.0f / HDIM) + EPSF);
    __syncthreads();

    qsh[d] = qv * rq_s; ksh[d] = kv * rk_s;
    __syncthreads();

    float qo, ko;
    const int j = d & 63;
    const float af = (j < 32) ? __powf(1.0f / 1024.0f, (float)j * (1.0f / 31.0f)) : 0.0f;
    const float th = (float)t * af;
    const float cc = cosf(th), ss = sinf(th);
    if (d < 64) { qo =  qsh[d] * cc + qsh[d + 64] * ss;  ko =  ksh[d] * cc + ksh[d + 64] * ss; }
    else        { qo = -qsh[j] * ss + qsh[d] * cc;        ko = -ksh[j] * ss + ksh[d] * cc; }

    const float lam0 = attn_lam[layer * 2 + 0];
    const float lam1 = attn_lam[layer * 2 + 1];
    float vo = lam0 * vv;
    if (vei >= 0)
        vo += lam1 * ve_w[((size_t)vei * VV + seq[t]) * DD + h * HDIM + d];

    const size_t o = ((size_t)h * TT + t) * HDIM + d;
    q16[o] = (_Float16)qo; k16[o] = (_Float16)ko; v16[o] = (_Float16)vo;
}

// =====================================================================
// 5) Block-sparse flash attention, shuffle-free softmax.
//    - flags for the whole q-row preloaded once (no per-iter global load)
//    - scores bounded => fixed-shift exp, no online max/rescale
//    - row sums via an extra WMMA against an all-ones V column tile
//    - K chunk staged via CDNA5 async global->LDS (ASYNCcnt)
// =====================================================================
__global__ __launch_bounds__(128)
void attn_kernel(const _Float16* __restrict__ q16, const _Float16* __restrict__ k16,
                 const _Float16* __restrict__ v16, _Float16* __restrict__ y16,
                 const int* __restrict__ docs, const unsigned char* __restrict__ bflags) {
    __shared__ _Float16 Ks[32 * 128];      // [key][d]
    __shared__ _Float16 Vs[128 * 32];      // [d][key] (transposed during staging)
    __shared__ int      sdocs[32];
    __shared__ _Float16 pls[4][16 * 32];   // per-wave P tile

    const int tid = threadIdx.x;
    const int wave = tid >> 5, lane = tid & 31;
    const int h = blockIdx.y;
    const int qt0 = blockIdx.x * 64 + wave * 16;
    const int qB = (blockIdx.x * 64) >> 7;           // uniform across waves
    const int khalf = lane >> 4;
    const int rowbase = khalf * 8;

    // preload the entire 16-byte flag row once (uniform address)
    unsigned flw[4];
    {
        const unsigned* fp = (const unsigned*)(bflags + qB * 16);
#pragma unroll
        for (int i = 0; i < 4; ++i) flw[i] = fp[i];
    }

    const _Float16* qh = q16 + (size_t)h * TT * HDIM;
    const _Float16* kh = k16 + (size_t)h * TT * HDIM;
    const _Float16* vh = v16 + (size_t)h * TT * HDIM;

    v16h aq[4];
    const int qrow = qt0 + (lane & 15);
#pragma unroll
    for (int dch = 0; dch < 4; ++dch)
        aq[dch] = frag16(qh + (size_t)qrow * HDIM + dch * 32 + khalf * 8,
                         qh + (size_t)qrow * HDIM + dch * 32 + 16 + khalf * 8);

    int qd[8];
#pragma unroll
    for (int r = 0; r < 8; ++r) qd[r] = docs[qt0 + rowbase + r];

    v16h ones;
#pragma unroll
    for (int i = 0; i < 16; ++i) ones[i] = (_Float16)1.0f;

    v8f oacc[8], osum;
#pragma unroll
    for (int nt = 0; nt < 8; ++nt)
#pragma unroll
        for (int r = 0; r < 8; ++r) oacc[nt][r] = 0.0f;
#pragma unroll
    for (int r = 0; r < 8; ++r) osum[r] = 0.0f;

    const int nkc_wave = (qt0 + 15) / 32 + 1;
    const int nkc_max  = blockIdx.x * 2 + 2;

    for (int kc = 0; kc < nkc_max; ++kc) {
        const int kB = kc >> 2;                       // (kc*32)>>7
        const unsigned flag = (flw[kB >> 2] >> ((kB & 3) * 8)) & 0xffu;
        if (flag == 0) continue;                      // uniform skip

        __syncthreads();
        {   // cooperative stage of 32 keys x 128 dims
            const int skey = tid >> 2;                // 0..31
            const int sd0  = (tid & 3) * 32;          // 0,32,64,96
            const _Float16* ksrc = kh + (size_t)(kc * 32 + skey) * HDIM + sd0;
            const _Float16* vsrc = vh + (size_t)(kc * 32 + skey) * HDIM + sd0;
            // K: async DMA straight into LDS (no VGPR round-trip)
#pragma unroll
            for (int u = 0; u < 4; ++u) {
                unsigned lofs = (unsigned)(size_t)&Ks[skey * 128 + sd0 + u * 8];
                asm volatile("global_load_async_to_lds_b128 %0, %1, off"
                             :: "v"(lofs), "v"(ksrc + u * 8) : "memory");
            }
            // V: load + transpose into [d][key]
#pragma unroll
            for (int u = 0; u < 4; ++u) {
                v8h vv = *(const v8h*)(vsrc + u * 8);
#pragma unroll
                for (int e = 0; e < 8; ++e)
                    Vs[(sd0 + u * 8 + e) * 32 + skey] = vv[e];
            }
            if (tid < 32) sdocs[tid] = docs[kc * 32 + tid];
            asm volatile("s_wait_asynccnt 0" ::: "memory");
        }
        __syncthreads();

        if (kc >= nkc_wave) continue;                 // staged for others only

#pragma unroll
        for (int jh = 0; jh < 2; ++jh) {
            v8f sc;
#pragma unroll
            for (int r = 0; r < 8; ++r) sc[r] = 0.0f;
            const int keyl = jh * 16 + (lane & 15);
#pragma unroll
            for (int dch = 0; dch < 4; ++dch) {
                const v16h bk = frag16(&Ks[keyl * 128 + dch * 32 + khalf * 8],
                                       &Ks[keyl * 128 + dch * 32 + 16 + khalf * 8]);
                sc = __builtin_amdgcn_wmma_f32_16x16x32_f16(
                    false, aq[dch], false, bk, (short)0, sc, false, false);
            }
#pragma unroll
            for (int r = 0; r < 8; ++r) {
                float sv = sc[r] * ATTN_SCALE;
                if (flag == 2) {
                    const int tq = qt0 + rowbase + r;
                    const int tk = kc * 32 + keyl;
                    if (tq < tk || qd[r] != sdocs[keyl]) sv = -1e30f;
                }
                pls[wave][(rowbase + r) * 32 + keyl] = (_Float16)__expf(sv - SMAX_SHIFT);
            }
        }

        // per-wave P transpose through LDS: hard-wait the DS counter
        asm volatile("s_wait_dscnt 0" ::: "memory");

        const int arow = lane & 15;
        const v16h ap = frag16(&pls[wave][arow * 32 + khalf * 8],
                               &pls[wave][arow * 32 + 16 + khalf * 8]);
#pragma unroll
        for (int nt = 0; nt < 8; ++nt) {
            const int col = nt * 16 + (lane & 15);
            const v16h bv = frag16(&Vs[col * 32 + khalf * 8],
                                   &Vs[col * 32 + 16 + khalf * 8]);
            oacc[nt] = __builtin_amdgcn_wmma_f32_16x16x32_f16(
                false, ap, false, bv, (short)0, oacc[nt], false, false);
        }
        // row sums of P: one WMMA against all-ones tile (no shuffles needed)
        osum = __builtin_amdgcn_wmma_f32_16x16x32_f16(
            false, ap, false, ones, (short)0, osum, false, false);
    }

#pragma unroll
    for (int nt = 0; nt < 8; ++nt) {
        const int d = nt * 16 + (lane & 15);
#pragma unroll
        for (int r = 0; r < 8; ++r) {
            const int tq = qt0 + rowbase + r;
            const float l = osum[r];
            const float o = (l > 0.0f) ? oacc[nt][r] / l : 0.0f;
            y16[(size_t)tq * DD + h * HDIM + d] = (_Float16)o;
        }
    }
}

// =====================================================================
// 6) Elementwise residual-stream helpers
// =====================================================================
__global__ void blocklam_kernel(float* __restrict__ x, const float* __restrict__ x0,
                                const float* __restrict__ block_lam, int layer) {
    const size_t i = (size_t)blockIdx.x * blockDim.x + threadIdx.x;
    x[i] = block_lam[layer * 2 + 0] * x[i] + block_lam[layer * 2 + 1] * x0[i];
}
__global__ void copy_kernel(float* __restrict__ dst, const float* __restrict__ src) {
    const size_t i = (size_t)blockIdx.x * blockDim.x + threadIdx.x;
    dst[i] = src[i];
}
__global__ void skipadd_kernel(float* __restrict__ x, const float* __restrict__ skip,
                               const float* __restrict__ skip_w, int j) {
    const size_t i = (size_t)blockIdx.x * blockDim.x + threadIdx.x;
    x[i] += skip_w[j] * skip[i];
}

// =====================================================================
// 7) Loss epilogue
// =====================================================================
__global__ void loss_init_kernel(float* __restrict__ rowsum, float* __restrict__ zt) {
    const int i = blockIdx.x * blockDim.x + threadIdx.x;
    if (i < TT) { rowsum[i] = 0.0f; zt[i] = 0.0f; }
}
__global__ void loss_final_kernel(const float* __restrict__ rowsum, const float* __restrict__ zt,
                                  float* __restrict__ out) {
    __shared__ float red[256];
    float s = 0.0f;
    for (int t = threadIdx.x; t < TT; t += 256) s += logf(rowsum[t]) - zt[t];
    red[threadIdx.x] = s; __syncthreads();
    for (int k = 128; k > 0; k >>= 1) { if (threadIdx.x < k) red[threadIdx.x] += red[threadIdx.x + k]; __syncthreads(); }
    if (threadIdx.x == 0) out[0] = red[0] / (float)TT;
}

// =====================================================================
// Host orchestration
// =====================================================================
extern "C" void kernel_launch(void* const* d_in, const int* in_sizes, int n_in,
                              void* d_out, int out_size, void* d_ws, size_t ws_size,
                              hipStream_t stream) {
    const int*   seq        = (const int*)d_in[0];
    const int*   tgt        = (const int*)d_in[1];
    const int*   swp        = (const int*)d_in[2];
    const float* embed_w    = (const float*)d_in[3];
    const float* ve_w       = (const float*)d_in[4];
    const float* qkv_w      = (const float*)d_in[5];
    const float* attn_lam   = (const float*)d_in[6];
    const float* attn_projw = (const float*)d_in[7];
    const float* mlp_fcw    = (const float*)d_in[8];
    const float* mlp_projw  = (const float*)d_in[9];
    const float* block_lam  = (const float*)d_in[10];
    const float* skip_w     = (const float*)d_in[11];
    const float* lm_head_w  = (const float*)d_in[12];

    char* p = (char*)d_ws;
    auto alloc = [&](size_t bytes) -> void* {
        void* r = (void*)p; p += (bytes + 255) & ~(size_t)255; return r;
    };
    float*     x      = (float*)alloc((size_t)TT * DD * 4);
    float*     x0     = (float*)alloc((size_t)TT * DD * 4);
    _Float16*  xn     = (_Float16*)alloc((size_t)TT * DD * 2);
    float*     qkv    = (float*)alloc((size_t)TT * 3 * DD * 4);
    _Float16*  q16    = (_Float16*)alloc((size_t)HH * TT * HDIM * 2);
    _Float16*  k16    = (_Float16*)alloc((size_t)HH * TT * HDIM * 2);
    _Float16*  v16    = (_Float16*)alloc((size_t)HH * TT * HDIM * 2);
    _Float16*  y16    = (_Float16*)alloc((size_t)TT * DD * 2);
    _Float16*  h16    = (_Float16*)alloc((size_t)TT * 4 * DD * 2);
    float*     skips  = (float*)alloc((size_t)6 * TT * DD * 4);
    int*       docs   = (int*)alloc((size_t)TT * 4);
    unsigned char* bflags = (unsigned char*)alloc(512);
    float*     rowsum = (float*)alloc((size_t)TT * 4);
    float*     ztag   = (float*)alloc((size_t)TT * 4);

    static const int mvar[12] = {0,1,1,1,0,1, 1,0,1,1,1,0};
    static const int veix[12] = {0,1,2,-1,-1,-1, -1,-1,-1,0,1,2};
    const float lm_inv = 1.0f / (7.5f * sqrtf((float)DD));
    const int NE = TT * DD;
    const dim3 ew((NE + 255) / 256), bt(256);

    mask_kernel<<<1, 32, 0, stream>>>(seq, swp, docs, bflags);
    embed_kernel<<<TT, 256, 0, stream>>>(seq, embed_w, x, x0);

    for (int i = 0; i < 12; ++i) {
        if (i >= 6) {
            const int j = i - 6;
            skipadd_kernel<<<ew, bt, 0, stream>>>(x, skips + (size_t)(5 - j) * NE, skip_w, j);
        }
        blocklam_kernel<<<ew, bt, 0, stream>>>(x, x0, block_lam, i);

        if (i != 7) {
            rmsnorm_f16_kernel<<<TT, 256, 0, stream>>>(x, xn);
            gemm_wmma<0><<<dim3(3 * DD / 64, TT / 64), 128, 0, stream>>>(
                xn, qkv_w + (size_t)i * 3 * DD * DD, qkv, nullptr, nullptr, nullptr, nullptr,
                TT, 3 * DD, DD, 0.0f);
            rope_kernel<<<dim3(TT, HH), HDIM, 0, stream>>>(
                qkv, attn_lam, i, ve_w, veix[i], seq, q16, k16, v16);
            attn_kernel<<<dim3(TT / 64, HH), 128, 0, stream>>>(
                q16, k16, v16, y16, docs, bflags + mvar[i] * 256);
            gemm_wmma<2><<<dim3(DD / 64, TT / 64), 128, 0, stream>>>(
                y16, attn_projw + (size_t)i * DD * DD, x, nullptr, nullptr, nullptr, nullptr,
                TT, DD, DD, 0.0f);
        }

        rmsnorm_f16_kernel<<<TT, 256, 0, stream>>>(x, xn);
        gemm_wmma<1><<<dim3(4 * DD / 64, TT / 64), 128, 0, stream>>>(
            xn, mlp_fcw + (size_t)i * 4 * DD * DD, nullptr, h16, nullptr, nullptr, nullptr,
            TT, 4 * DD, DD, 0.0f);
        gemm_wmma<2><<<dim3(DD / 64, TT / 64), 128, 0, stream>>>(
            h16, mlp_projw + (size_t)i * DD * 4 * DD, x, nullptr, nullptr, nullptr, nullptr,
            TT, DD, 4 * DD, 0.0f);

        if (i < 6)
            copy_kernel<<<ew, bt, 0, stream>>>(skips + (size_t)i * NE, x);
    }

    rmsnorm_f16_kernel<<<TT, 256, 0, stream>>>(x, xn);
    loss_init_kernel<<<(TT + 255) / 256, 256, 0, stream>>>(rowsum, ztag);
    gemm_wmma<3><<<dim3(VPAD / 64, TT / 64), 128, 0, stream>>>(
        xn, lm_head_w, nullptr, nullptr, rowsum, ztag, tgt, TT, VPAD, DD, lm_inv);
    loss_final_kernel<<<1, 256, 0, stream>>>(rowsum, ztag, (float*)d_out);
}
